// Glimpse_71107478553000
// MI455X (gfx1250) — compile-verified, matrix operand order checked by reading, not run on hardware
//
#include <hip/hip_runtime.h>
#include <cstdint>
#include <cstddef>

// ---- CDNA5 wave32 WMMA types ----
typedef __attribute__((ext_vector_type(16))) __bf16  v16bf;
typedef __attribute__((ext_vector_type(8)))  float   v8f;
// POD vector types (HIP's float4/ushort4 have ctors -> not union-safe)
typedef __attribute__((ext_vector_type(4)))  float          f32x4;
typedef __attribute__((ext_vector_type(4)))  unsigned int   u32x4;
typedef __attribute__((ext_vector_type(4)))  unsigned short u16x4;

// ds_swizzle group-of-32: offset = xor_mask<<10 | or_mask<<5 | and_mask(0x1f)
#define SWZF(x, imm) __int_as_float(__builtin_amdgcn_ds_swizzle(__float_as_int(x), (imm)))
#define SWZ_X16 0x401F   // xor 16: exchange half-waves

__device__ __forceinline__ unsigned short f2bf(float f) {
  unsigned u = __float_as_uint(f);
  u += 0x7FFFu + ((u >> 16) & 1u);   // round-to-nearest-even truncation
  return (unsigned short)(u >> 16);
}

union Frag {                 // one WMMA 16-bit A/B operand (8 VGPRs)
  v16bf          v;
  unsigned short s[16];
  u32x4          q[2];
};

constexpr int BATCH = 256;
constexpr int NNODE = 2048;
constexpr int EMB   = 128;
constexpr int HEADS = 8;
constexpr int DK    = 16;
constexpr int SPLIT = 4;                 // N-axis split -> 1024 workgroups
constexpr int CHUNK = NNODE / SPLIT;     // 512 nodes per block
constexpr int TILE  = 128;               // node rows staged per block iteration
constexpr int NPAD  = EMB + 8;           // padded LDS row (bf16 elems) - bank spread
// ws layout per (b,h,split): 16 acc + m + l  = 18 floats  (total ~576 KB)

__global__ __launch_bounds__(256)
void glimpse_attn_partial(const float* __restrict__ ctx,
                          const float* __restrict__ node,
                          const unsigned char* __restrict__ mask,
                          const float* __restrict__ Wq,
                          const float* __restrict__ bq,
                          const float* __restrict__ Wk,
                          const float* __restrict__ bk,
                          const float* __restrict__ Wv,
                          float* __restrict__ ws)
{
  __shared__ __attribute__((aligned(16))) unsigned short tileA[TILE * NPAD];
  __shared__ __attribute__((aligned(16))) float sAddT[TILE];  // mask -> -2e6 / 0
  __shared__ float sQ[HEADS * DK];        // scaled q per head
  __shared__ float sWQ[HEADS][EMB];       // q-folded score weights per head

  const int b    = blockIdx.x;
  const int sp   = blockIdx.y;
  const int tid  = threadIdx.x;
  const int lane = tid & 31;
  const int h    = tid >> 5;     // wave index == head index
  const int j    = lane & 15;    // N column within 16-wide tile / d-index
  const int hi   = lane >> 4;    // half-wave selects K/M sub-range

  const float* nodeB        = node + (size_t)b * NNODE * EMB;
  const unsigned char* mskB = mask + (size_t)b * NNODE;

  // ---- Wv B-fragments for this head, kept in registers (bf16) ----
  // B layout: lane(j,hi) holds column N=j, K = kb*32 + hi*8 + {0..7, 16..23}
  // == contiguous runs of row-major weight row (16h+j).
  Frag fv[4];
  const int wrow = h * DK + j;
#pragma unroll
  for (int kb = 0; kb < 4; ++kb) {
    const float* pv = Wv + (size_t)wrow * EMB + kb * 32 + hi * 8;
#pragma unroll
    for (int i = 0; i < 8; ++i) {
      fv[kb].s[i]     = f2bf(pv[i]);
      fv[kb].s[8 + i] = f2bf(pv[16 + i]);
    }
  }

  // ---- q for this head, pre-scaled by 1/sqrt(DK)=0.25; lane j computes q_j ----
  {
    float qv = bq[wrow];
    const float* cb = ctx + (size_t)b * EMB;
    const float* wq = Wq  + (size_t)wrow * EMB;
#pragma unroll 4
    for (int e = 0; e < EMB; ++e) qv += cb[e] * wq[e];
    if (lane < DK) sQ[h * DK + j] = qv * 0.25f;   // intra-wave LDS (dscnt waits)
  }

  // qb = sum_j q_j * bk_j  (per-head scalar, identical across lanes)
  float qb = 0.0f;
#pragma unroll
  for (int jj = 0; jj < DK; ++jj) qb += sQ[h * DK + jj] * bk[h * DK + jj];

  // wq_eff[e] = sum_j q_j * Wk[16h+j][e]  (cooperative across the wave, coalesced)
#pragma unroll
  for (int t = 0; t < EMB / 32; ++t) {
    int e = lane + t * 32;
    float s = 0.0f;
#pragma unroll
    for (int jj = 0; jj < DK; ++jj)
      s += sQ[h * DK + jj] * Wk[(size_t)(h * DK + jj) * EMB + e];
    sWQ[h][e] = s;
  }

  // Bq fragments: ALL 16 columns = wq_eff  =>  score WMMA yields D[m][j]=s[m]
  // replicated across lanes (exactly the layout the softmax / p.V step needs).
  Frag fq[4];
#pragma unroll
  for (int kb = 0; kb < 4; ++kb) {
#pragma unroll
    for (int i = 0; i < 8; ++i) {
      int e0 = kb * 32 + hi * 8 + i;
      fq[kb].s[i]     = f2bf(sWQ[h][e0]);
      fq[kb].s[8 + i] = f2bf(sWQ[h][e0 + 16]);
    }
  }

  float m_run = -1e30f, l_run = 0.0f, acc = 0.0f;  // acc: out[d=j], dup per half

  const int n0 = sp * CHUNK;
  for (int t0 = n0; t0 < n0 + CHUNK; t0 += TILE) {
    __syncthreads();                    // prior tile fully consumed
    // stage 128x128 node rows as bf16 into LDS (coalesced f32x4 reads)
#pragma unroll
    for (int i = 0; i < (TILE * EMB / 4) / 256; ++i) {
      int e   = (tid + i * 256) << 2;
      int row = e >> 7;
      int col = e & (EMB - 1);
      f32x4 f = *(const f32x4*)(nodeB + (size_t)(t0 + row) * EMB + col);
      u16x4 h4 = { f2bf(f.x), f2bf(f.y), f2bf(f.z), f2bf(f.w) };
      *(u16x4*)&tileA[row * NPAD + col] = h4;
    }
    if (tid < TILE) sAddT[tid] = mskB[t0 + tid] ? -2000000.0f : 0.0f;
    __syncthreads();

    // prefetch next tile into cache while we compute on this one
    if (t0 + TILE < n0 + CHUNK) {
#pragma unroll
      for (int i = 0; i < (TILE * EMB / 4) / 256; ++i) {
        int e   = (tid + i * 256) << 2;
        int row = e >> 7;
        int col = e & (EMB - 1);
        __builtin_prefetch(nodeB + (size_t)(t0 + TILE + row) * EMB + col, 0, 3);
      }
    }

#pragma unroll 1
    for (int mt = 0; mt < TILE / 16; ++mt) {
      // A-fragments: lane(j,hi) = row mt*16+j, K = kb*32 + hi*8 + {0..7,16..23}
      Frag a[4];
      const int arow = mt * 16 + j;
#pragma unroll
      for (int kb = 0; kb < 4; ++kb) {
        const unsigned short* p = &tileA[arow * NPAD + kb * 32 + hi * 8];
        a[kb].q[0] = *(const u32x4*)p;
        a[kb].q[1] = *(const u32x4*)(p + 16);
      }

      // scores directly from WMMA: ck[r] = s_raw[m = r + 8*hi] (all lanes)
      v8f ck = {};
#pragma unroll
      for (int kb = 0; kb < 4; ++kb)
        ck = __builtin_amdgcn_wmma_f32_16x16x32_bf16(false, a[kb].v, false, fq[kb].v,
                                                     (short)0, ck, false, false);

      // mask via clamped addend: masked rows become exactly -1e6
      f32x4 m0 = *(const f32x4*)&sAddT[mt * 16 + hi * 8];
      f32x4 m1 = *(const f32x4*)&sAddT[mt * 16 + hi * 8 + 4];
      float sr[8];
#pragma unroll
      for (int r = 0; r < 4; ++r) sr[r]     = fmaxf(ck[r]     + qb + m0[r], -1000000.0f);
#pragma unroll
      for (int r = 0; r < 4; ++r) sr[4 + r] = fmaxf(ck[4 + r] + qb + m1[r], -1000000.0f);

      // online softmax update
      float tm = sr[0];
#pragma unroll
      for (int r = 1; r < 8; ++r) tm = fmaxf(tm, sr[r]);
      tm = fmaxf(tm, SWZF(tm, SWZ_X16));
      float mn   = fmaxf(m_run, tm);
      float corr = __expf(m_run - mn);
      float ts = 0.0f;
#pragma unroll
      for (int r = 0; r < 8; ++r) { sr[r] = __expf(sr[r] - mn); ts += sr[r]; }
      ts += SWZF(ts, SWZ_X16);
      l_run = l_run * corr + ts;

      // V-projection tile + accumulate: acc[j] += sum_m p_m * v[m][j]
      v8f cv = {};
#pragma unroll
      for (int kb = 0; kb < 4; ++kb)
        cv = __builtin_amdgcn_wmma_f32_16x16x32_bf16(false, a[kb].v, false, fv[kb].v,
                                                     (short)0, cv, false, false);
      float pv = 0.0f;
#pragma unroll
      for (int r = 0; r < 8; ++r) pv += sr[r] * cv[r];  // rows of this half
      pv += SWZF(pv, SWZ_X16);                          // add other half's rows
      acc = acc * corr + pv;
      m_run = mn;
    }
  }

  float* w = ws + ((size_t)(b * HEADS + h) * SPLIT + sp) * 18;
  if (lane < 16) w[j] = acc;
  if (lane == 0) { w[16] = m_run; w[17] = l_run; }
}

__global__ __launch_bounds__(256)
void glimpse_attn_combine(const float* __restrict__ ws,
                          const float* __restrict__ bv,
                          float* __restrict__ out)
{
  int idx = blockIdx.x * blockDim.x + threadIdx.x;   // (b*H + h)*16 + d
  if (idx >= BATCH * HEADS * DK) return;
  int d  = idx & (DK - 1);
  int bh = idx >> 4;
  int hh = bh & (HEADS - 1);
  const float* base = ws + (size_t)bh * SPLIT * 18;
  float m = base[16];
#pragma unroll
  for (int s = 1; s < SPLIT; ++s) m = fmaxf(m, base[s * 18 + 16]);
  float l = 0.0f, a = 0.0f;
#pragma unroll
  for (int s = 0; s < SPLIT; ++s) {
    float c = __expf(base[s * 18 + 16] - m);
    l += base[s * 18 + 17] * c;
    a += base[s * 18 + d] * c;
  }
  out[idx] = a / l + bv[hh * DK + d];   // bv folds in exactly (sum p / l == 1)
}

extern "C" void kernel_launch(void* const* d_in, const int* in_sizes, int n_in,
                              void* d_out, int out_size, void* d_ws, size_t ws_size,
                              hipStream_t stream) {
  const float*         ctx  = (const float*)d_in[0];
  const float*         node = (const float*)d_in[1];
  const unsigned char* mask = (const unsigned char*)d_in[2];  // jax bool = 1 byte
  const float*         Wq   = (const float*)d_in[3];
  const float*         bq   = (const float*)d_in[4];
  const float*         Wk   = (const float*)d_in[5];
  const float*         bk   = (const float*)d_in[6];
  const float*         Wv   = (const float*)d_in[7];
  const float*         bv   = (const float*)d_in[8];
  float* out = (float*)d_out;
  float* ws  = (float*)d_ws;   // needs 256*8*4*18*4 = 589,824 bytes

  dim3 grid1(BATCH, SPLIT);
  glimpse_attn_partial<<<grid1, 256, 0, stream>>>(ctx, node, mask, Wq, bq, Wk, bk, Wv, ws);
  int total = BATCH * HEADS * DK;
  glimpse_attn_combine<<<(total + 255) / 256, 256, 0, stream>>>(ws, bv, out);
}
